// MLPomega_79113297592867
// MI455X (gfx1250) — matrix operand order repaired and checked by reading, not compiled, not run on hardware
//
#include <hip/hip_runtime.h>
#include <hip/hip_bf16.h>
#include <cfloat>

// ---------------------------------------------------------------------------
// Problem constants (reference: B=128, N=256, D=512, O=512)
// ---------------------------------------------------------------------------
#define BATCH   128
#define SEQ     256
#define DIM     512            // D == O == 512
#define ROWS    (BATCH * SEQ)  // 32768
#define MTILE   64             // rows per workgroup
#define NBLOCKS (ROWS / MTILE) // 512
#define NTHREADS 512           // 16 wave32s
#define KTILES  (DIM / 32)     // 16 k-steps of 32
#define NTILES  (DIM / 16)     // 32 n-tiles of 16
// packed weight: per matrix 512 tiles * 32 lanes * 8 dwords = 131072 dwords
#define WPACK_DWORDS_PER_MAT (KTILES * NTILES * 32 * 8)

typedef _Float16 v16h __attribute__((ext_vector_type(16)));
typedef float    v8f  __attribute__((ext_vector_type(8)));

union FragAB { uint4 u[2]; v16h h; };

// ---------------------------------------------------------------------------
// Kernel 1: convert fp32 weights -> f16 packed in WMMA B-fragment layout.
// B tile (32x16, KxN): lane half g holds K=8g..8g+7 and 16+8g..16+8g+7 (N=lane&15);
// dword i of a lane packs halves (K=8g+2(i&3)+16(i>>2), +1).
// Dest dword index == flat thread id by construction.
// ---------------------------------------------------------------------------
__global__ void pack_weights(const float* __restrict__ W1,
                             const float* __restrict__ W2,
                             const float* __restrict__ W3,
                             unsigned int* __restrict__ Wp) {
  int t = blockIdx.x * blockDim.x + threadIdx.x;
  if (t >= 3 * WPACK_DWORDS_PER_MAT) return;
  int i    = t & 7;            // dword within fragment
  int lane = (t >> 3) & 31;
  int tile = (t >> 8) & 511;   // kt*32 + nt
  int w    = t >> 17;          // which weight matrix
  int nt = tile & 31;
  int kt = tile >> 5;
  int g  = lane >> 4;
  int n  = nt * 16 + (lane & 15);
  int k0 = kt * 32 + 8 * g + 2 * (i & 3) + 16 * (i >> 2);
  const float* W = (w == 0) ? W1 : (w == 1) ? W2 : W3;
  union { _Float16 h[2]; unsigned int u; } p;
  p.h[0] = (_Float16)W[(size_t)k0 * DIM + n];
  p.h[1] = (_Float16)W[(size_t)(k0 + 1) * DIM + n];
  Wp[t] = p.u;
}

// ---------------------------------------------------------------------------
// Kernel 2: per-batch "any(mask)" for the at-least-one-valid fixup
// ---------------------------------------------------------------------------
__global__ void any_mask(const int* __restrict__ mask, int* __restrict__ anyM) {
  __shared__ int flag;
  if (threadIdx.x == 0) flag = 0;
  __syncthreads();
  if (mask[(size_t)blockIdx.x * SEQ + threadIdx.x] != 0) atomicOr(&flag, 1);
  __syncthreads();
  if (threadIdx.x == 0) anyM[blockIdx.x] = flag;
}

// ---------------------------------------------------------------------------
// Fused MLP helpers
// ---------------------------------------------------------------------------
__device__ __forceinline__ float wave_reduce_add(float v) {
  v += __shfl_xor(v, 16, 32);
  v += __shfl_xor(v, 8, 32);
  v += __shfl_xor(v, 4, 32);
  v += __shfl_xor(v, 2, 32);
  v += __shfl_xor(v, 1, 32);
  return v;
}

// LayerNorm 4 rows/wave from a strided fp32 source into f16 sA (row-major).
template <typename SrcFn>
__device__ __forceinline__ void ln_rows(SrcFn src,
                                        const float* __restrict__ gam,
                                        const float* __restrict__ bet,
                                        _Float16* __restrict__ sA,
                                        int wave, int lane) {
  for (int rr = 0; rr < 4; ++rr) {
    int row = wave * 4 + rr;
    float vals[16];
    float s = 0.f;
#pragma unroll
    for (int j = 0; j < 16; ++j) { vals[j] = src(row, j * 32 + lane); s += vals[j]; }
    s = wave_reduce_add(s);
    float mean = s * (1.0f / 512.0f);
    float vs = 0.f;
#pragma unroll
    for (int j = 0; j < 16; ++j) { float d = vals[j] - mean; vs += d * d; }
    vs = wave_reduce_add(vs);
    float rstd = rsqrtf(vs * (1.0f / 512.0f) + 1e-5f);
#pragma unroll
    for (int j = 0; j < 16; ++j) {
      int col = j * 32 + lane;
      sA[row * DIM + col] = (_Float16)((vals[j] - mean) * rstd * gam[col] + bet[col]);
    }
  }
}

// One 64x512 = f16(sA 64x512) x f16(Wp 512x512) GEMM, f32 accumulate.
// Wave w owns all 4 M-tiles and N-tiles {2w, 2w+1}: each B fragment is
// loaded exactly once per block (weights stream from L2).
__device__ __forceinline__ void gemm_block(const _Float16* __restrict__ sA,
                                           const unsigned int* __restrict__ Wp,
                                           int wave, int lane, v8f acc[4][2]) {
  const int g  = lane >> 4;
  const int ml = lane & 15;
  for (int kt = 0; kt < KTILES; ++kt) {
    FragAB a[4];
#pragma unroll
    for (int mt = 0; mt < 4; ++mt) {
      // A 16x32 f16 fragment: lane (g,ml) -> row mt*16+ml,
      // dwords 0..3 = K kt*32+8g+{0..7}, dwords 4..7 = +16. Two ds_load_b128.
      const _Float16* p = sA + (mt * 16 + ml) * DIM + kt * 32 + 8 * g;
      a[mt].u[0] = *(const uint4*)p;
      a[mt].u[1] = *(const uint4*)(p + 16);
    }
    FragAB b[2];
#pragma unroll
    for (int j = 0; j < 2; ++j) {
      int nt = wave * 2 + j;
      const uint4* wp = (const uint4*)(Wp + (((size_t)kt * NTILES + nt) * 32 + lane) * 8);
      b[j].u[0] = wp[0];
      b[j].u[1] = wp[1];
    }
#pragma unroll
    for (int mt = 0; mt < 4; ++mt)
#pragma unroll
      for (int j = 0; j < 2; ++j)
        acc[mt][j] = __builtin_amdgcn_wmma_f32_16x16x32_f16(
            /*neg_a=*/false, a[mt].h, /*neg_b=*/false, b[j].h,
            /*c_mod=*/(short)0, acc[mt][j], /*reuse_a=*/false, /*reuse_b=*/false);
  }
}

// ---------------------------------------------------------------------------
// Kernel 3: fused LN1->GEMM1->tanh->LN2->GEMM2->tanh+res->LN3->GEMM3->mask
// Block = 64 rows, 512 threads (16 waves). LDS: sA 64KB f16 + sH 128KB f32.
// ---------------------------------------------------------------------------
__global__ void __launch_bounds__(NTHREADS)
fused_mlp(const float* __restrict__ X,
          const int* __restrict__ mask,
          const int* __restrict__ anyM,
          const float* __restrict__ g1, const float* __restrict__ b1,
          const float* __restrict__ g2, const float* __restrict__ b2,
          const float* __restrict__ g3, const float* __restrict__ b3,
          const unsigned int* __restrict__ Wp,
          float* __restrict__ Out) {
  extern __shared__ char smem[];
  _Float16* sA   = (_Float16*)smem;                                  // 64*512 f16
  float*    sH   = (float*)(smem + MTILE * DIM * 2);                 // 64*512 f32
  int*      sMsk = (int*)(smem + MTILE * DIM * 2 + MTILE * DIM * 4); // 64 ints

  const int tid  = threadIdx.x;
  const int wave = tid >> 5;
  const int lane = tid & 31;
  const int g    = lane >> 4;
  const int ml   = lane & 15;
  const size_t r0 = (size_t)blockIdx.x * MTILE;

  if (tid < MTILE) {
    size_t gr = r0 + tid;
    int bb = (int)(gr >> 8);               // SEQ == 256
    int valid = (mask[gr] != 0);
    if (!valid && ((gr & 255) == 0) && (anyM[bb] == 0)) valid = 1;
    sMsk[tid] = valid;
  }

  // ---- Block 1: LN(x) -> sA (f16) ----
  ln_rows([&](int row, int col) { return X[(r0 + row) * DIM + col]; },
          g1, b1, sA, wave, lane);
  __syncthreads();

  const v8f vzero = {0.f, 0.f, 0.f, 0.f, 0.f, 0.f, 0.f, 0.f};
  v8f acc[4][2];

  // ---- GEMM1 + tanh -> sH (h1) ----
#pragma unroll
  for (int mt = 0; mt < 4; ++mt)
    for (int j = 0; j < 2; ++j) acc[mt][j] = vzero;
  gemm_block(sA, Wp, wave, lane, acc);
#pragma unroll
  for (int mt = 0; mt < 4; ++mt)
#pragma unroll
    for (int j = 0; j < 2; ++j) {
      int n = (wave * 2 + j) * 16 + ml;
#pragma unroll
      for (int vi = 0; vi < 8; ++vi) {
        int m = mt * 16 + 8 * g + vi;
        sH[m * DIM + n] = tanhf(acc[mt][j][vi]);
      }
    }
  __syncthreads();

  // ---- Block 2: LN(h1) -> sA ----
  ln_rows([&](int row, int col) { return sH[row * DIM + col]; },
          g2, b2, sA, wave, lane);
  __syncthreads();

  // ---- GEMM2 + tanh + residual(h1) -> sH ----
#pragma unroll
  for (int mt = 0; mt < 4; ++mt)
    for (int j = 0; j < 2; ++j) acc[mt][j] = vzero;
  gemm_block(sA, Wp + WPACK_DWORDS_PER_MAT, wave, lane, acc);
#pragma unroll
  for (int mt = 0; mt < 4; ++mt)
#pragma unroll
    for (int j = 0; j < 2; ++j) {
      int n = (wave * 2 + j) * 16 + ml;
#pragma unroll
      for (int vi = 0; vi < 8; ++vi) {
        int m = mt * 16 + 8 * g + vi;
        float r = tanhf(acc[mt][j][vi]) + sH[m * DIM + n];  // h2 + h1
        sH[m * DIM + n] = r;                                 // unique owner
      }
    }
  __syncthreads();

  // ---- Block 3: LN(h2+h1) -> sA ----
  ln_rows([&](int row, int col) { return sH[row * DIM + col]; },
          g3, b3, sA, wave, lane);
  __syncthreads();

  // ---- GEMM3, mask fill, write logits ----
#pragma unroll
  for (int mt = 0; mt < 4; ++mt)
    for (int j = 0; j < 2; ++j) acc[mt][j] = vzero;
  gemm_block(sA, Wp + 2 * WPACK_DWORDS_PER_MAT, wave, lane, acc);
#pragma unroll
  for (int mt = 0; mt < 4; ++mt)
#pragma unroll
    for (int j = 0; j < 2; ++j) {
      int n = (wave * 2 + j) * 16 + ml;
#pragma unroll
      for (int vi = 0; vi < 8; ++vi) {
        int m = mt * 16 + 8 * g + vi;
        size_t gr = r0 + m;
        float v = sMsk[m] ? acc[mt][j][vi] : -FLT_MAX;  // jnp.finfo(f32).min
        Out[gr * DIM + n] = v;
      }
    }
}

// ---------------------------------------------------------------------------
// Kernel 4: softmax over axis=1 (the N/sequence axis), in place on d_out.
// One thread per (b, o) column; threads with consecutive o coalesce, and the
// 64MB logit buffer is fully L2-resident (192MB) for the re-read passes.
// ---------------------------------------------------------------------------
__global__ void softmax_n(float* __restrict__ Out) {
  int c = blockIdx.x * blockDim.x + threadIdx.x;  // 0 .. B*DIM-1
  int b = c >> 9;             // /512
  int o = c & (DIM - 1);
  float* base = Out + ((size_t)b * SEQ * DIM + o);
  float mx = -FLT_MAX;
  for (int n = 0; n < SEQ; ++n) mx = fmaxf(mx, base[(size_t)n * DIM]);
  float s = 0.f;
  for (int n = 0; n < SEQ; ++n) s += expf(base[(size_t)n * DIM] - mx);
  float inv = 1.0f / s;
  for (int n = 0; n < SEQ; ++n) {
    float e = expf(base[(size_t)n * DIM] - mx) * inv;  // exp(-huge)=0 for masked
    base[(size_t)n * DIM] = e;
  }
}

// ---------------------------------------------------------------------------
extern "C" void kernel_launch(void* const* d_in, const int* in_sizes, int n_in,
                              void* d_out, int out_size, void* d_ws, size_t ws_size,
                              hipStream_t stream) {
  (void)in_sizes; (void)n_in; (void)out_size; (void)ws_size;
  const float* X  = (const float*)d_in[0];
  const int*   M  = (const int*)d_in[1];   // bool mask as int32 per harness convention
  const float* g1 = (const float*)d_in[2];
  const float* b1 = (const float*)d_in[3];
  const float* W1 = (const float*)d_in[4];
  const float* g2 = (const float*)d_in[5];
  const float* b2 = (const float*)d_in[6];
  const float* W2 = (const float*)d_in[7];
  const float* g3 = (const float*)d_in[8];
  const float* b3 = (const float*)d_in[9];
  const float* W3 = (const float*)d_in[10];
  float* Out = (float*)d_out;

  unsigned int* Wp   = (unsigned int*)d_ws;                       // 1.5 MB packed f16 weights
  int*          anyM = (int*)((char*)d_ws + (size_t)3 * WPACK_DWORDS_PER_MAT * 4);

  const size_t smem = (size_t)MTILE * DIM * 2   // sA f16
                    + (size_t)MTILE * DIM * 4   // sH f32
                    + (size_t)MTILE * 4;        // sMask
  // Opt in to >64KB dynamic LDS (WGP has 320KB). Deterministic, capture-safe.
  (void)hipFuncSetAttribute((const void*)fused_mlp,
                            hipFuncAttributeMaxDynamicSharedMemorySize, (int)smem);

  pack_weights<<<(3 * WPACK_DWORDS_PER_MAT + 255) / 256, 256, 0, stream>>>(W1, W2, W3, Wp);
  any_mask<<<BATCH, SEQ, 0, stream>>>(M, anyM);
  fused_mlp<<<NBLOCKS, NTHREADS, smem, stream>>>(X, M, anyM, g1, b1, g2, b2, g3, b3, Wp, Out);
  softmax_n<<<(BATCH * DIM) / 256, 256, 0, stream>>>(Out);
}
